// GATBlock_23819888624110
// MI455X (gfx1250) — compile-verified, matrix operand order checked by reading, not run on hardware
//
#include <hip/hip_runtime.h>
#include <hip/hip_bf16.h>
#include <stdint.h>

typedef __attribute__((ext_vector_type(16))) __bf16 v16bf;
typedef __attribute__((ext_vector_type(8)))  float  v8f;

#define HEADS 8
#define NEG_SLOPE 0.2f

// ---------------- utility kernels ----------------

__global__ void k_build_edges(const int* __restrict__ ei, int* __restrict__ src,
                              int* __restrict__ dst, int E, int Nn) {
  int i = blockIdx.x * blockDim.x + threadIdx.x;
  int etot = E + Nn;
  if (i >= etot) return;
  if (i < E) { src[i] = ei[i]; dst[i] = ei[E + i]; }
  else       { src[i] = i - E; dst[i] = i - E; }      // self loops
}

__global__ void k_cvt_bf16(const float* __restrict__ in, __bf16* __restrict__ out, int n) {
  int i = blockIdx.x * blockDim.x + threadIdx.x;
  if (i < n) out[i] = (__bf16)in[i];
}

__global__ void k_fill(float* __restrict__ p, float v, int n) {
  int i = blockIdx.x * blockDim.x + threadIdx.x;
  if (i < n) p[i] = v;
}

// ---------------- WMMA GEMM: D[M,Nc] = A[M,K](bf16) * W[Nc,K](bf16)^T ----------------
// 128 threads = 4 waves stacked in M. Block tile 128x64; wave tile 32x64; K-step 32.
// Per wave per K-step: 2 A-frags + 4 B-frags (12 ds_load_b128) feed 8 WMMAs.
__global__ __launch_bounds__(128)
void k_wmma_gemm(const __bf16* __restrict__ A, const __bf16* __restrict__ W,
                 float* __restrict__ D, int M, int Nc, int K) {
  __shared__ __align__(16) unsigned int As[128 * 20]; // 128 rows x 32 bf16, stride 20 dwords
  __shared__ __align__(16) unsigned int Bs[64 * 20];  // 64 rows x 32 bf16
  const int tid  = threadIdx.x;
  const int wave = tid >> 5;
  const int lane = tid & 31;
  const int l    = lane & 15;
  const int hi   = lane >> 4;
  const int row0 = blockIdx.y * 128;
  const int col0 = blockIdx.x * 64;
  const unsigned int* Au = (const unsigned int*)A;
  const unsigned int* Wu = (const unsigned int*)W;
  const int Kd = K >> 1;                 // dwords per row

  const int r    = tid >> 1;             // 0..63
  const int half = tid & 1;              // which 8-dword half of a 32-elem row

  v8f acc[2][4];
  v8f vzero = {0.f, 0.f, 0.f, 0.f, 0.f, 0.f, 0.f, 0.f};
#pragma unroll
  for (int mi = 0; mi < 2; ++mi)
#pragma unroll
    for (int j = 0; j < 4; ++j) acc[mi][j] = vzero;

  for (int k0 = 0; k0 < K; k0 += 32) {
    const int kd0 = k0 >> 1;
    // stage A tile: rows r and r+64 (guard M)
#pragma unroll
    for (int t = 0; t < 2; ++t) {
      int rr = r + t * 64;
      uint4 q0 = make_uint4(0u, 0u, 0u, 0u), q1 = q0;
      int grow = row0 + rr;
      if (grow < M) {
        const uint4* ap = (const uint4*)(Au + (size_t)grow * Kd + kd0) + half * 2;
        q0 = ap[0]; q1 = ap[1];
      }
      uint4* lp = (uint4*)&As[rr * 20 + half * 8];
      lp[0] = q0; lp[1] = q1;
    }
    // stage B tile (Nc is a multiple of 64, always in range)
    {
      const uint4* wp = (const uint4*)(Wu + (size_t)(col0 + r) * Kd + kd0) + half * 2;
      uint4* lp = (uint4*)&Bs[r * 20 + half * 8];
      lp[0] = wp[0]; lp[1] = wp[1];
    }
    __syncthreads();

    // fragments: lanes 0-15 hold K pairs {0..3, 8..11}; lanes 16-31 hold {4..7, 12..15}
    union Frag { uint4 q[2]; v16bf v; };
    Frag af[2];
#pragma unroll
    for (int mi = 0; mi < 2; ++mi) {
      const unsigned int* arow = &As[(wave * 32 + mi * 16 + l) * 20 + hi * 4];
      af[mi].q[0] = *(const uint4*)arow;
      af[mi].q[1] = *(const uint4*)(arow + 8);
    }
#pragma unroll
    for (int j = 0; j < 4; ++j) {
      Frag bfm;
      const unsigned int* brow = &Bs[(j * 16 + l) * 20 + hi * 4];
      bfm.q[0] = *(const uint4*)brow;
      bfm.q[1] = *(const uint4*)(brow + 8);
#pragma unroll
      for (int mi = 0; mi < 2; ++mi)
        acc[mi][j] = __builtin_amdgcn_wmma_f32_16x16x32_bf16(false, af[mi].v, false, bfm.v,
                                                             (short)0, acc[mi][j], false, false);
    }
    __syncthreads();
  }

  // C/D layout: VGPR rr -> M = rr + 8*hi, N = l
#pragma unroll
  for (int mi = 0; mi < 2; ++mi) {
#pragma unroll
    for (int j = 0; j < 4; ++j) {
#pragma unroll
      for (int rr = 0; rr < 8; ++rr) {
        int row = row0 + wave * 32 + mi * 16 + hi * 8 + rr;
        if (row < M) D[(size_t)row * Nc + (col0 + j * 16 + l)] = acc[mi][j][rr];
      }
    }
  }
}

// ---------------- attention logits: per (node, head) dot over C ----------------
__global__ __launch_bounds__(256)
void k_alpha(const float* __restrict__ Hf, const float* __restrict__ a_s,
             const float* __restrict__ a_d, float* __restrict__ AS,
             float* __restrict__ AD, int nH, int Nc, int C) {
  int wid  = blockIdx.x * (blockDim.x >> 5) + (threadIdx.x >> 5);
  int lane = threadIdx.x & 31;
  if (wid >= nH) return;
  int node = wid >> 3;
  int h    = wid & 7;
  const float* hp = Hf + (size_t)node * Nc + h * C;
  const float* sp = a_s + h * C;
  const float* dp = a_d + h * C;
  float ss = 0.f, sd = 0.f;
  for (int c = lane; c < C; c += 32) { float v = hp[c]; ss += v * sp[c]; sd += v * dp[c]; }
#pragma unroll
  for (int o = 16; o > 0; o >>= 1) { ss += __shfl_xor(ss, o, 32); sd += __shfl_xor(sd, o, 32); }
  if (lane == 0) { AS[wid] = ss; AD[wid] = sd; }
}

__device__ __forceinline__ float leaky(float v) { return v > 0.f ? v : NEG_SLOPE * v; }

__device__ __forceinline__ float atomicMaxF(float* addr, float val) {
  if (val >= 0.f) return __int_as_float(atomicMax((int*)addr, __float_as_int(val)));
  return __uint_as_float(atomicMin((unsigned int*)addr, __float_as_uint(val)));
}

__global__ void k_edge_max(const int* __restrict__ src, const int* __restrict__ dst,
                           const float* __restrict__ AS, const float* __restrict__ AD,
                           float* __restrict__ MX, int eth) {
  int i = blockIdx.x * blockDim.x + threadIdx.x;
  if (i >= eth) return;
  int e = i >> 3, h = i & 7;
  int s = src[e], d = dst[e];
  float v = leaky(AS[s * HEADS + h] + AD[d * HEADS + h]);
  atomicMaxF(&MX[d * HEADS + h], v);
}

__global__ void k_edge_exp(const int* __restrict__ src, const int* __restrict__ dst,
                           const float* __restrict__ AS, const float* __restrict__ AD,
                           const float* __restrict__ MX, float* __restrict__ EH,
                           float* __restrict__ SM, int eth) {
  int i = blockIdx.x * blockDim.x + threadIdx.x;
  if (i >= eth) return;
  int e = i >> 3, h = i & 7;
  int s = src[e], d = dst[e];
  float v = leaky(AS[s * HEADS + h] + AD[d * HEADS + h]);
  float p = __expf(v - MX[d * HEADS + h]);
  EH[i] = p;
  atomicAdd(&SM[d * HEADS + h], p);
}

// wave per (edge, head); lanes stride the C channels; fp32 atomic scatter
__global__ __launch_bounds__(256)
void k_aggregate(const int* __restrict__ src, const int* __restrict__ dst,
                 const float* __restrict__ EH, const float* __restrict__ SM,
                 const float* __restrict__ Hf, float* __restrict__ OUT,
                 int Nc, int C, int eth) {
  int wid = blockIdx.x * (blockDim.x >> 5) + (threadIdx.x >> 5);
  if (wid >= eth) return;
  int lane = threadIdx.x & 31;
  int e = wid >> 3, h = wid & 7;
  int s = src[e], d = dst[e];
  float alpha = EH[wid] / (SM[d * HEADS + h] + 1e-16f);
  const float* hp = Hf + (size_t)s * Nc + h * C;
  float*       op = OUT + (size_t)d * Nc + h * C;
  for (int c = lane; c < C; c += 32)
    atomicAdd(&op[c], alpha * hp[c]);
}

__global__ void k_bias_elu(const float* __restrict__ OUT, const float* __restrict__ b,
                           __bf16* __restrict__ XB, int Nc, int total) {
  int i = blockIdx.x * blockDim.x + threadIdx.x;
  if (i >= total) return;
  float v = OUT[i] + b[i & (Nc - 1)];
  v = v > 0.f ? v : (__expf(v) - 1.f);
  XB[i] = (__bf16)v;
}

__global__ void k_head_mean(const float* __restrict__ OUT, const float* __restrict__ b,
                            float* __restrict__ out, int C, int total) {
  int i = blockIdx.x * blockDim.x + threadIdx.x;
  if (i >= total) return;
  int node = i / C, c = i - node * C;
  float ssum = 0.f;
#pragma unroll
  for (int h = 0; h < HEADS; ++h)
    ssum += OUT[(size_t)node * (HEADS * C) + h * C + c];
  out[i] = ssum * (1.0f / HEADS) + b[c];
}

// ---------------- host launcher ----------------

extern "C" void kernel_launch(void* const* d_in, const int* in_sizes, int n_in,
                              void* d_out, int out_size, void* d_ws, size_t ws_size,
                              hipStream_t stream) {
  (void)n_in; (void)out_size; (void)ws_size;
  const float* x  = (const float*)d_in[0];
  const int*   ei = (const int*)d_in[1];
  const float* Wm[3]  = {(const float*)d_in[2], (const float*)d_in[6],  (const float*)d_in[10]};
  const float* asv[3] = {(const float*)d_in[3], (const float*)d_in[7],  (const float*)d_in[11]};
  const float* adv[3] = {(const float*)d_in[4], (const float*)d_in[8],  (const float*)d_in[12]};
  const float* bv[3]  = {(const float*)d_in[5], (const float*)d_in[9],  (const float*)d_in[13]};

  const int IN   = 256;
  const int n    = in_sizes[0] / IN;          // 10000
  const int E    = in_sizes[1] / 2;           // 160000
  const int etot = E + n;                     // + self loops
  const int Kdim[3] = {256, 512, 1024};
  const int Ncol[3] = {512, 1024, 2048};
  const int Chan[3] = {64, 128, 256};

  char* base = (char*)d_ws;
  size_t off = 0;
  auto carve = [&](size_t bytes) -> void* {
    void* p = base + off;
    off += (bytes + 255) & ~(size_t)255;
    return p;
  };
  int*    src = (int*)carve((size_t)etot * 4);
  int*    dst = (int*)carve((size_t)etot * 4);
  __bf16* XB  = (__bf16*)carve((size_t)n * 1024 * 2);     // bf16 activations (max in-dim 1024)
  __bf16* WB  = (__bf16*)carve((size_t)2048 * 1024 * 2);  // bf16 weights (max W3)
  float*  HF  = (float*)carve((size_t)n * 2048 * 4);      // transformed features
  float*  OUT = (float*)carve((size_t)n * 2048 * 4);      // aggregated output
  float*  AS  = (float*)carve((size_t)n * HEADS * 4);
  float*  AD  = (float*)carve((size_t)n * HEADS * 4);
  float*  MX  = (float*)carve((size_t)n * HEADS * 4);
  float*  SM  = (float*)carve((size_t)n * HEADS * 4);
  float*  EH  = (float*)carve((size_t)etot * HEADS * 4);

  auto cdiv = [](int a, int b) { return (a + b - 1) / b; };

  k_build_edges<<<cdiv(etot, 256), 256, 0, stream>>>(ei, src, dst, E, n);
  k_cvt_bf16<<<cdiv(n * IN, 256), 256, 0, stream>>>(x, XB, n * IN);

  for (int L = 0; L < 3; ++L) {
    const int K = Kdim[L], Nc = Ncol[L], C = Chan[L];
    k_cvt_bf16<<<cdiv(Nc * K, 256), 256, 0, stream>>>(Wm[L], WB, Nc * K);

    dim3 g(Nc / 64, cdiv(n, 128));
    k_wmma_gemm<<<g, 128, 0, stream>>>(XB, WB, HF, n, Nc, K);

    const int nH = n * HEADS;
    k_alpha<<<cdiv(nH, 8), 256, 0, stream>>>(HF, asv[L], adv[L], AS, AD, nH, Nc, C);

    k_fill<<<cdiv(nH, 256), 256, 0, stream>>>(MX, -3.0e38f, nH);
    hipMemsetAsync(SM, 0, (size_t)nH * 4, stream);
    hipMemsetAsync(OUT, 0, (size_t)n * Nc * 4, stream);

    const int eth = etot * HEADS;
    k_edge_max<<<cdiv(eth, 256), 256, 0, stream>>>(src, dst, AS, AD, MX, eth);
    k_edge_exp<<<cdiv(eth, 256), 256, 0, stream>>>(src, dst, AS, AD, MX, EH, SM, eth);
    k_aggregate<<<cdiv(eth, 8), 256, 0, stream>>>(src, dst, EH, SM, HF, OUT, Nc, C, eth);

    if (L < 2) {
      k_bias_elu<<<cdiv(n * Nc, 256), 256, 0, stream>>>(OUT, bv[L], XB, Nc, n * Nc);
    } else {
      k_head_mean<<<cdiv(n * C, 256), 256, 0, stream>>>(OUT, bv[2], (float*)d_out, C, n * C);
    }
  }
}
